// RAFT_19628000543450
// MI455X (gfx1250) — matrix operand order repaired and checked by reading, not compiled, not run on hardware
//
#include <hip/hip_runtime.h>
#include <math.h>

typedef _Float16 h16;
typedef __attribute__((ext_vector_type(16))) _Float16 v16h;
typedef __attribute__((ext_vector_type(8)))  _Float16 v8h;
typedef __attribute__((ext_vector_type(2)))  _Float16 v2h;
typedef __attribute__((ext_vector_type(8)))  float    v8f;
typedef int v4i __attribute__((vector_size(16)));   // matches builtin param type

#define ACT_NONE 0
#define ACT_RELU 1
#define ACT_SIG  2
#define ACT_TANH 3

static const long BCOL_CAP = 22000000L;   // elements of f16 im2col scratch

#if defined(__gfx1250__) && __has_builtin(__builtin_amdgcn_global_load_async_to_lds_b128) && __has_builtin(__builtin_amdgcn_s_wait_asynccnt)
#define HAS_ASYNC_LDS 1
#else
#define HAS_ASYNC_LDS 0
#endif

// ------------------------------------------------------------------ kernels

__global__ void k_convert(const float* __restrict__ s, h16* __restrict__ d, long n) {
  long i = (long)blockIdx.x * blockDim.x + threadIdx.x;
  if (i < n) d[i] = (h16)s[i];
}

__global__ void k_im2col(const float* __restrict__ in, h16* __restrict__ out,
                         int Cin, int H, int W, int KH, int KW, int stride,
                         int ph, int pw, int Ho, int Wo,
                         long inBS, long outBS, long total) {
  long idx = (long)blockIdx.x * blockDim.x + threadIdx.x;
  if (idx >= total) return;
  long per = (long)Cin * KH * KW * Ho * Wo;
  int b = (int)(idx / per);
  long r0 = idx - (long)b * per;
  int HoWo = Ho * Wo;
  int k = (int)(r0 / HoWo);
  int p = (int)(r0 - (long)k * HoWo);
  int ci = k / (KH * KW);
  int rs = k - ci * (KH * KW);
  int r = rs / KW, s = rs - r * KW;
  int ho = p / Wo, wo = p - ho * Wo;
  int hi = ho * stride - ph + r;
  int wi = wo * stride - pw + s;
  float v = 0.f;
  if (hi >= 0 && hi < H && wi >= 0 && wi < W)
    v = in[(long)b * inBS + ((long)ci * H + hi) * W + wi];
  out[(long)b * outBS + r0] = (h16)v;
}

// C[M,N] = act( scale * A[M,K] x B[K,N] + bias[M] ), f16 operands, f32 accum.
// dst[( (batchBase+z)*dstCtot + coOff + m ) * N + n]
__launch_bounds__(128)
__global__ void k_gemm(const h16* __restrict__ A, const h16* __restrict__ B,
                       const float* __restrict__ bias, float* __restrict__ D,
                       int M, int N, int K, long aS, long bS,
                       int dstCtot, int coOff, int batchBase,
                       int act, float scale) {
  __shared__ __align__(32) h16 As[64 * 32];
  __shared__ __align__(32) h16 Bs[64 * 32];

  const h16* Ab = A + (long)blockIdx.z * aS;
  const h16* Bb = B + (long)blockIdx.z * bS;
  int mBase = blockIdx.y * 64, nBase = blockIdx.x * 64;
  int tid = threadIdx.x, lane = tid & 31, wid = tid >> 5;

  // block-uniform interior-tile predicates (scalar branches, no exec juggling)
  const bool fullM = (mBase + 64 <= M) && ((K & 7) == 0);
  const bool fullN = (nBase + 64 <= N) && ((N & 7) == 0);

  // A-fill mapping: row = tid/4 (+32), 8 contiguous k at (tid%4)*8
  int ar = tid >> 2;
  int ac = (tid & 3) * 8;
  // B-fill fast mapping: k-pair rows x 8 contiguous n
  int bk2 = (tid & 15) * 2;
  int bn0 = (tid >> 4) * 8;
  // B-fill slow mapping
  int skk = tid & 31;
  int sn0 = (tid >> 5) * 16;

  v8f acc[4];
#pragma unroll
  for (int j = 0; j < 4; ++j)
#pragma unroll
    for (int e = 0; e < 8; ++e) acc[j][e] = 0.0f;

  for (int k0 = 0; k0 < K; k0 += 32) {
    bool fullK = (k0 + 32 <= K);
    bool aFast = fullM && fullK;

    // -------- A tile: LDS layout identical to global (row-major 64x32)
    if (aFast) {
#if HAS_ASYNC_LDS
      h16* g0 = const_cast<h16*>(Ab) + (long)(mBase + ar) * K + k0 + ac;
      h16* g1 = const_cast<h16*>(Ab) + (long)(mBase + ar + 32) * K + k0 + ac;
      __builtin_amdgcn_global_load_async_to_lds_b128(
          (__attribute__((address_space(1))) v4i*)(uintptr_t)g0,
          (__attribute__((address_space(3))) v4i*)&As[ar * 32 + ac], 0, 0);
      __builtin_amdgcn_global_load_async_to_lds_b128(
          (__attribute__((address_space(1))) v4i*)(uintptr_t)g1,
          (__attribute__((address_space(3))) v4i*)&As[(ar + 32) * 32 + ac], 0, 0);
#else
      *(v8h*)&As[ar * 32 + ac] = *(const v8h*)(Ab + (long)(mBase + ar) * K + k0 + ac);
      *(v8h*)&As[(ar + 32) * 32 + ac] = *(const v8h*)(Ab + (long)(mBase + ar + 32) * K + k0 + ac);
#endif
    } else {
#pragma unroll
      for (int pass = 0; pass < 2; ++pass) {
        int r = ar + pass * 32;
        int gm = mBase + r;
#pragma unroll
        for (int i = 0; i < 8; ++i) {
          int kk = k0 + ac + i;
          h16 v = (h16)0.f;
          if (gm < M && kk < K) v = Ab[(long)gm * K + kk];
          As[r * 32 + ac + i] = v;
        }
      }
    }

    // -------- B tile transposed into Bs[n][k]
    if (fullN && fullK) {
      const h16* r0p = Bb + (long)(k0 + bk2) * N + nBase + bn0;
      v8h lo = *(const v8h*)r0p;
      v8h hi = *(const v8h*)(r0p + N);
#pragma unroll
      for (int i = 0; i < 8; ++i) {
        v2h pr;
        pr[0] = lo[i];
        pr[1] = hi[i];
        *(v2h*)&Bs[(bn0 + i) * 32 + bk2] = pr;   // ds_store_b32 pair store
      }
    } else {
      int gk = k0 + skk;
#pragma unroll
      for (int i = 0; i < 16; ++i) {
        int gn = nBase + sn0 + i;
        h16 v = (h16)0.f;
        if (gk < K && gn < N) v = Bb[(long)gk * N + gn];
        Bs[(sn0 + i) * 32 + skk] = v;
      }
    }

    if (k0 + 32 < K)  // pull next B tile toward L2 (global_prefetch_b8)
      __builtin_prefetch(&Bb[(long)(k0 + 32 + skk) * N + nBase + sn0], 0, 1);

#if HAS_ASYNC_LDS
    if (aFast) __builtin_amdgcn_s_wait_asynccnt(0);
#endif
    __syncthreads();

    // -------- WMMA: each wave computes a 16x64 strip
    int halfSel = lane >> 4;            // ISA A-operand lane split
    int arow = wid * 16 + (lane & 15);
    v8h alo = *(const v8h*)&As[arow * 32 + halfSel * 16];
    v8h ahi = *(const v8h*)&As[arow * 32 + halfSel * 16 + 8];
    v16h af;
#pragma unroll
    for (int i = 0; i < 8; ++i) { af[i] = alo[i]; af[8 + i] = ahi[i]; }
#pragma unroll
    for (int j = 0; j < 4; ++j) {
      v16h bf = *(const v16h*)&Bs[(j * 16 + (lane & 15)) * 32 + halfSel * 16];
      acc[j] = __builtin_amdgcn_wmma_f32_16x16x32_f16(
          false, af, false, bf, (short)0, acc[j], false, false);
    }
    __syncthreads();
  }

  // -------- store with fused bias/activation
  int rowSel = 8 * (lane >> 4), colSel = lane & 15;
  long dBase = (long)((batchBase + blockIdx.z) * dstCtot + coOff);
  if (mBase + 64 <= M && nBase + 64 <= N) {   // interior: unguarded stores
#pragma unroll
    for (int j = 0; j < 4; ++j) {
      int n = nBase + j * 16 + colSel;
#pragma unroll
      for (int v = 0; v < 8; ++v) {
        int m = mBase + wid * 16 + rowSel + v;
        float val = acc[j][v] * scale;
        if (bias) val += bias[m];
        if (act == ACT_RELU) val = fmaxf(val, 0.f);
        else if (act == ACT_SIG) val = 1.f / (1.f + expf(-val));
        else if (act == ACT_TANH) val = tanhf(val);
        D[(dBase + m) * N + n] = val;
      }
    }
  } else {
#pragma unroll
    for (int j = 0; j < 4; ++j) {
      int n = nBase + j * 16 + colSel;
      if (n >= N) continue;
#pragma unroll
      for (int v = 0; v < 8; ++v) {
        int m = mBase + wid * 16 + rowSel + v;
        if (m >= M) continue;
        float val = acc[j][v] * scale;
        if (bias) val += bias[m];
        if (act == ACT_RELU) val = fmaxf(val, 0.f);
        else if (act == ACT_SIG) val = 1.f / (1.f + expf(-val));
        else if (act == ACT_TANH) val = tanhf(val);
        D[(dBase + m) * N + n] = val;
      }
    }
  }
}

// mode 0: instance norm (biased var); mode 1: BN-eval scale. One block per (n,c).
__global__ void k_norm(const float* __restrict__ x, float* __restrict__ y,
                       int HW, int mode, int doRelu) {
  __shared__ float ss[256], sq[256];
  long base = (long)blockIdx.x * HW;
  int tid = threadIdx.x;
  if (mode == 1) {
    float sc = rsqrtf(1.0f + 1e-5f);
    for (int i = tid; i < HW; i += 256) {
      float v = x[base + i] * sc;
      if (doRelu) v = fmaxf(v, 0.f);
      y[base + i] = v;
    }
    return;
  }
  float s = 0.f, q = 0.f;
  for (int i = tid; i < HW; i += 256) {
    float v = x[base + i];
    s += v; q += v * v;
  }
  ss[tid] = s; sq[tid] = q;
  __syncthreads();
  for (int o = 128; o > 0; o >>= 1) {
    if (tid < o) { ss[tid] += ss[tid + o]; sq[tid] += sq[tid + o]; }
    __syncthreads();
  }
  float mean = ss[0] / HW;
  float var = sq[0] / HW - mean * mean;
  float inv = rsqrtf(fmaxf(var, 0.f) + 1e-5f);
  for (int i = tid; i < HW; i += 256) {
    float v = (x[base + i] - mean) * inv;
    if (doRelu) v = fmaxf(v, 0.f);
    y[base + i] = v;
  }
}

__global__ void k_addrelu(const float* __restrict__ a, const float* __restrict__ b,
                          float* __restrict__ o, long n) {
  long i = (long)blockIdx.x * blockDim.x + threadIdx.x;
  if (i < n) o[i] = fmaxf(a[i] + b[i], 0.f);
}

__global__ void k_addinp(float* __restrict__ a, const float* __restrict__ b, long n) {
  long i = (long)blockIdx.x * blockDim.x + threadIdx.x;
  if (i < n) a[i] += b[i];
}

__global__ void k_gruupd(const float* __restrict__ z, const float* __restrict__ q,
                         float* __restrict__ net, long n) {
  long i = (long)blockIdx.x * blockDim.x + threadIdx.x;
  if (i < n) net[i] = (1.f - z[i]) * net[i] + z[i] * q[i];
}

__global__ void k_copych(float* __restrict__ dst, int dstCtot, int dstOff,
                         const float* __restrict__ src, int srcCtot, int srcOff,
                         int C, int HW, int nb) {
  long i = (long)blockIdx.x * blockDim.x + threadIdx.x;
  long total = (long)nb * C * HW;
  if (i >= total) return;
  int q = (int)(i % HW);
  long t = i / HW;
  int c = (int)(t % C);
  int b = (int)(t / C);
  dst[((long)b * dstCtot + dstOff + c) * HW + q] =
      src[((long)b * srcCtot + srcOff + c) * HW + q];
}

__global__ void k_rmul(const float* __restrict__ r, const float* __restrict__ net,
                       float* __restrict__ hx, int HW, int nb) {
  long i = (long)blockIdx.x * blockDim.x + threadIdx.x;
  long total = (long)nb * 128 * HW;
  if (i >= total) return;
  int q = (int)(i % HW);
  long t = i / HW;
  int c = (int)(t % 128);
  int b = (int)(t / 128);
  hx[((long)b * 384 + c) * HW + q] = r[i] * net[i];
}

__global__ void k_flow(const float* __restrict__ coords, float* __restrict__ flowb,
                       float* __restrict__ hx, int nb, int h, int w) {
  int hw = h * w;
  long i = (long)blockIdx.x * blockDim.x + threadIdx.x;
  long total = (long)nb * 2 * hw;
  if (i >= total) return;
  int q = (int)(i % hw);
  long t = i / hw;
  int ch = (int)(t % 2);
  int b = (int)(t / 2);
  float g = (ch == 0) ? (float)(q % w) : (float)(q / w);
  float v = coords[((long)b * 2 + ch) * hw + q] - g;
  flowb[i] = v;
  if (hx) hx[((long)b * 384 + 382 + ch) * hw + q] = v;
}

__global__ void k_initcoords(float* __restrict__ coords, int nb, int h, int w) {
  int hw = h * w;
  long i = (long)blockIdx.x * blockDim.x + threadIdx.x;
  long total = (long)nb * 2 * hw;
  if (i >= total) return;
  int q = (int)(i % hw);
  int ch = (int)((i / hw) % 2);
  coords[i] = (ch == 0) ? (float)(q % w) : (float)(q / w);
}

__global__ void k_pool2(const float* __restrict__ in, float* __restrict__ out,
                        int N, int h, int w) {
  int ho = h >> 1, wo = w >> 1;
  long i = (long)blockIdx.x * blockDim.x + threadIdx.x;
  long total = (long)N * ho * wo;
  if (i >= total) return;
  int xo = (int)(i % wo);
  long t = i / wo;
  int yo = (int)(t % ho);
  int n = (int)(t / ho);
  const float* p = in + ((long)n * h + 2 * yo) * w + 2 * xo;
  out[i] = 0.25f * (p[0] + p[1] + p[w] + p[w + 1]);
}

// fmaps [4][256][2240] -> At[b][q][c] (f16, GEMM-A for f1), Bc[b][c][q] (f16, GEMM-B for f2)
__global__ void k_fmap2ab(const float* __restrict__ fm, h16* __restrict__ At,
                          h16* __restrict__ Bc, int C, int HW) {
  long i = (long)blockIdx.x * blockDim.x + threadIdx.x;
  long total = 2L * C * HW;
  if (i >= total) return;
  int q = (int)(i % HW);
  long t = i / HW;
  int c = (int)(t % C);
  int b = (int)(t / C);
  At[((long)b * HW + q) * C + c] = (h16)fm[((long)b * C + c) * HW + q];
  Bc[((long)b * C + c) * HW + q] = (h16)fm[((long)(b + 2) * C + c) * HW + q];
}

__device__ __forceinline__ float bilin_tap(const float* img, int hh, int ww, int yy, int xx) {
  if (xx < 0 || xx > ww - 1 || yy < 0 || yy > hh - 1) return 0.f;
  return img[yy * ww + xx];
}

// 4-level, 81-tap correlation lookup -> corrfeat f16 [b][324][hw] (direct GEMM-B for menc.c1)
__global__ void k_sampler(const float* __restrict__ coords,
                          const float* __restrict__ p0, const float* __restrict__ p1,
                          const float* __restrict__ p2, const float* __restrict__ p3,
                          h16* __restrict__ out, int B, int h, int w) {
  int hw = h * w;
  long idx = (long)blockIdx.x * blockDim.x + threadIdx.x;
  long total = (long)B * hw * 4 * 81;
  if (idx >= total) return;
  int j = (int)(idx % 81);
  long t = idx / 81;
  int lvl = (int)(t % 4); t /= 4;
  int q = (int)(t % hw);
  int b = (int)(t / hw);
  float cx = coords[((long)b * 2 + 0) * hw + q];
  float cy = coords[((long)b * 2 + 1) * hw + q];
  float s = 1.0f / (float)(1 << lvl);
  float dx = (float)(j / 9) - 4.0f;
  float dy = (float)(j % 9) - 4.0f;
  float x = cx * s + dx, y = cy * s + dy;
  int hh = h >> lvl, ww = w >> lvl;
  const float* vol = (lvl == 0) ? p0 : (lvl == 1) ? p1 : (lvl == 2) ? p2 : p3;
  const float* img = vol + (long)(b * hw + q) * hh * ww;
  float x0f = floorf(x), y0f = floorf(y);
  float wx = x - x0f, wy = y - y0f;
  int x0 = (int)x0f, y0 = (int)y0f;
  float v = (1.f - wy) * (1.f - wx) * bilin_tap(img, hh, ww, y0, x0)
          + (1.f - wy) * wx * bilin_tap(img, hh, ww, y0, x0 + 1)
          + wy * (1.f - wx) * bilin_tap(img, hh, ww, y0 + 1, x0)
          + wy * wx * bilin_tap(img, hh, ww, y0 + 1, x0 + 1);
  out[((long)(b * 324) + lvl * 81 + j) * hw + q] = (h16)v;
}

// convex 8x upsample: thread per (b,h,w,i,j), both channels
__global__ void k_upsample(const float* __restrict__ flow, const float* __restrict__ mask,
                           float* __restrict__ out, int B, int h, int w) {
  int hw = h * w;
  long idx = (long)blockIdx.x * blockDim.x + threadIdx.x;
  long total = (long)B * hw * 64;
  if (idx >= total) return;
  int ij = (int)(idx % 64);
  long t = idx / 64;
  int ww_ = (int)(t % w);
  long t2 = t / w;
  int hh = (int)(t2 % h);
  int b = (int)(t2 / h);
  int i = ij / 8, j = ij % 8;
  float m[9], mx = -1e30f;
#pragma unroll
  for (int k = 0; k < 9; ++k) {
    m[k] = mask[((long)(b * 576) + (k * 64 + ij)) * hw + hh * w + ww_];
    mx = fmaxf(mx, m[k]);
  }
  float sum = 0.f;
#pragma unroll
  for (int k = 0; k < 9; ++k) { m[k] = expf(m[k] - mx); sum += m[k]; }
  float inv = 1.f / sum;
#pragma unroll
  for (int ch = 0; ch < 2; ++ch) {
    float acc = 0.f;
#pragma unroll
    for (int k = 0; k < 9; ++k) {
      int hi = hh + k / 3 - 1, wi = ww_ + k % 3 - 1;
      float f = 0.f;
      if (hi >= 0 && hi < h && wi >= 0 && wi < w)
        f = flow[((long)(b * 2 + ch) * h + hi) * w + wi];
      acc += m[k] * inv * 8.f * f;
    }
    out[((long)(b * 2 + ch) * (8 * h) + (hh * 8 + i)) * (8 * w) + (ww_ * 8 + j)] = acc;
  }
}

// ------------------------------------------------------------------ host side

static inline void launch_im2col(hipStream_t st, const float* in, long inBS, int nb,
                                 int Cin, int H, int W, int KH, int KW, int stride,
                                 int ph, int pw, int Ho, int Wo, h16* out, long outBS) {
  long total = (long)nb * Cin * KH * KW * Ho * Wo;
  unsigned blocks = (unsigned)((total + 255) / 256);
  k_im2col<<<blocks, 256, 0, st>>>(in, out, Cin, H, W, KH, KW, stride, ph, pw, Ho, Wo,
                                   inBS, outBS, total);
}

static inline void launch_gemm(hipStream_t st, const h16* A, const h16* B,
                               const float* bias, float* D, int M, int N, int K,
                               long aS, long bS, int dstCtot, int coOff, int batchBase,
                               int nb, int act, float scale) {
  dim3 g((N + 63) / 64, (M + 63) / 64, nb);
  k_gemm<<<g, 128, 0, st>>>(A, B, bias, D, M, N, K, aS, bS, dstCtot, coOff, batchBase,
                            act, scale);
}

static void conv2d(hipStream_t st, const float* in, long inBS, int nb,
                   int Cin, int H, int W, const h16* Wgt, const float* bias,
                   int Cout, int KH, int KW, int stride, int ph, int pw,
                   float* dst, int dstCtot, int coOff, int batchBase,
                   int act, float scale, h16* Bcol) {
  int Ho = (H + 2 * ph - KH) / stride + 1;
  int Wo = (W + 2 * pw - KW) / stride + 1;
  int K = Cin * KH * KW, N = Ho * Wo;
  long tile = (long)K * N;
  if ((long)nb * tile <= BCOL_CAP) {
    launch_im2col(st, in, inBS, nb, Cin, H, W, KH, KW, stride, ph, pw, Ho, Wo, Bcol, tile);
    launch_gemm(st, Wgt, Bcol, bias, dst, Cout, N, K, 0, tile, dstCtot, coOff, batchBase,
                nb, act, scale);
  } else {
    for (int b = 0; b < nb; ++b) {
      launch_im2col(st, in + (long)b * inBS, 0, 1, Cin, H, W, KH, KW, stride, ph, pw,
                    Ho, Wo, Bcol, 0);
      launch_gemm(st, Wgt, Bcol, bias, dst, Cout, N, K, 0, 0, dstCtot, coOff,
                  batchBase + b, 1, act, scale);
    }
  }
}

static inline void launch_norm(hipStream_t st, const float* x, float* y, int NC, int HW,
                               int mode, int relu) {
  k_norm<<<NC, 256, 0, st>>>(x, y, HW, mode, relu);
}
static inline unsigned eb(long n) { return (unsigned)((n + 255) / 256); }

struct EncIdx {
  int conv1, conv2b, conv2w;
  int l1r1c1, l1r1c2, l1r2c1, l1r2c2;
  int l2r1c1, l2r1c2, l2r1d, l2r2c1, l2r2c2;
  int l3r1c1, l3r1c2, l3r1d, l3r2c1, l3r2c2;
};

extern "C" void kernel_launch(void* const* d_in, const int* in_sizes, int n_in,
                              void* d_out, int out_size, void* d_ws, size_t ws_size,
                              hipStream_t stream) {
  if (n_in < 66) return;
  const int Bn = 2, H = 320, W = 448, hh = 40, wwq = 56, HW = hh * wwq;   // HW=2240
  const int ITERS = 12;

  // ---- param index map (JAX pytree sorted-key flatten order) ----
  EncIdx C = {2, 3, 4, 5, 6, 7, 8, 9, 10, 11, 12, 13, 14, 15, 16, 17, 18};
  EncIdx F = {23, 24, 25, 26, 27, 28, 29, 30, 31, 32, 33, 34, 35, 36, 37, 38, 39};
  const int Gq1b = 40, Gq1w = 41, Gq2b = 42, Gq2w = 43, Gr1b = 44, Gr1w = 45,
            Gr2b = 46, Gr2w = 47, Gz1b = 48, Gz1w = 49, Gz2b = 50, Gz2w = 51;
  const int Mm1b = 52, Mm1w = 53, Mm2b = 54, Mm2w = 55;
  const int Ec1b = 56, Ec1w = 57, Ec2b = 58, Ec2w = 59, Ecb = 60, Ecw = 61,
            Ef1b = 62, Ef1w = 63, Ef2b = 64, Ef2w = 65;
  const int Fc1b = 19, Fc1w = 20, Fc2b = 21, Fc2w = 22;

  // ---- workspace bump allocator ----
  char* base = (char*)d_ws;
  size_t off = 0;
  auto aF = [&](long e) { float* p = (float*)(base + off); off += (((size_t)e * 4 + 255) / 256) * 256; return p; };
  auto aH = [&](long e) { h16* p = (h16*)(base + off); off += (((size_t)e * 2 + 255) / 256) * 256; return p; };

  float* bufA = aF(4L * 64 * 160 * 224);
  float* bufB = aF(4L * 64 * 160 * 224);
  float* bufC = aF(4L * 64 * 160 * 224);
  float* fmap = aF(4L * 256 * HW);
  float* corr0 = aF(2L * HW * HW);
  float* corr1 = aF(2L * HW * 560);
  float* corr2 = aF(2L * HW * 140);
  float* corr3 = aF(2L * HW * 35);
  float* net   = aF(2L * 128 * HW);
  float* hx    = aF(2L * 384 * HW);
  float* zb    = aF(2L * 128 * HW);
  float* rb_   = aF(2L * 128 * HW);
  float* qb    = aF(2L * 128 * HW);
  float* cor1  = aF(2L * 256 * HW);
  float* mencat= aF(2L * 256 * HW);
  float* flo1  = aF(2L * 128 * HW);
  float* fh1   = aF(2L * 256 * HW);
  float* maskb = aF(2L * 576 * HW);
  float* coords= aF(2L * 2 * HW);
  float* flowb = aF(2L * 2 * HW);
  float* dflow = aF(2L * 2 * HW);

  h16* Bcol = aH(BCOL_CAP);
  h16* At   = aH(2L * HW * 256);
  h16* Bc   = aH(2L * 256 * HW);
  h16* cfeat= aH(2L * 324 * HW);

  // ---- convert all weight leaves to f16 ----
  size_t totP = 0;
  for (int i = 2; i < n_in; ++i) totP += (size_t)in_sizes[i];
  h16* wpool = aH((long)totP);
  const h16* wh[66];
  {
    size_t o = 0;
    for (int i = 2; i < n_in && i < 66; ++i) {
      h16* p = wpool + o;
      o += (size_t)in_sizes[i];
      k_convert<<<eb(in_sizes[i]), 256, 0, stream>>>((const float*)d_in[i], p, in_sizes[i]);
      wh[i] = p;
    }
  }
  auto bia = [&](int i) { return (const float*)d_in[i]; };

  const float* image1 = (const float*)d_in[0];
  const float* image2 = (const float*)d_in[1];

  // ---- shared encoder ----
  auto run_encoder = [&](const EncIdx& E, const float* i1, const float* i2, int nb,
                         int mode) -> float* {
    conv2d(stream, i1, 3L * H * W, 2, 3, H, W, wh[E.conv1], nullptr, 64, 7, 7, 2, 3, 3,
           bufA, 64, 0, 0, ACT_NONE, 1.f, Bcol);
    if (i2)
      conv2d(stream, i2, 3L * H * W, 2, 3, H, W, wh[E.conv1], nullptr, 64, 7, 7, 2, 3, 3,
             bufA, 64, 0, 2, ACT_NONE, 1.f, Bcol);
    launch_norm(stream, bufA, bufA, nb * 64, 160 * 224, mode, 1);
    float* cur = bufA; float* s1 = bufB; float* s2 = bufC;
    auto rblk = [&](int c1, int c2, int dn, int Cin, int Cout, int Hh, int Ww, int st2) {
      int Ho = Hh / st2, Wo = Ww / st2, HWo = Ho * Wo;
      conv2d(stream, cur, (long)Cin * Hh * Ww, nb, Cin, Hh, Ww, wh[c1], nullptr, Cout,
             3, 3, st2, 1, 1, s1, Cout, 0, 0, ACT_NONE, 1.f, Bcol);
      launch_norm(stream, s1, s1, nb * Cout, HWo, mode, 1);
      conv2d(stream, s1, (long)Cout * HWo, nb, Cout, Ho, Wo, wh[c2], nullptr, Cout,
             3, 3, 1, 1, 1, s1, Cout, 0, 0, ACT_NONE, 1.f, Bcol);
      launch_norm(stream, s1, s1, nb * Cout, HWo, mode, 1);
      if (dn >= 0) {
        conv2d(stream, cur, (long)Cin * Hh * Ww, nb, Cin, Hh, Ww, wh[dn], nullptr, Cout,
               1, 1, st2, 0, 0, s2, Cout, 0, 0, ACT_NONE, 1.f, Bcol);
        launch_norm(stream, s2, s2, nb * Cout, HWo, mode, 0);
        long n = (long)nb * Cout * HWo;
        k_addrelu<<<eb(n), 256, 0, stream>>>(s2, s1, s2, n);
        float* old = cur; cur = s2; s2 = old;
      } else {
        long n = (long)nb * Cout * HWo;
        k_addrelu<<<eb(n), 256, 0, stream>>>(cur, s1, cur, n);
      }
    };
    rblk(E.l1r1c1, E.l1r1c2, -1, 64, 64, 160, 224, 1);
    rblk(E.l1r2c1, E.l1r2c2, -1, 64, 64, 160, 224, 1);
    rblk(E.l2r1c1, E.l2r1c2, E.l2r1d, 64, 96, 160, 224, 2);
    rblk(E.l2r2c1, E.l2r2c2, -1, 96, 96, 80, 112, 1);
    rblk(E.l3r1c1, E.l3r1c2, E.l3r1d, 96, 128, 80, 112, 2);
    rblk(E.l3r2c1, E.l3r2c2, -1, 128, 128, 40, 56, 1);
    return cur;
  };

  // fnet on concat(images) (batch 4), instance norm
  float* curF = run_encoder(F, image1, image2, 4, 0);
  conv2d(stream, curF, 128L * HW, 4, 128, hh, wwq, wh[F.conv2w], bia(F.conv2b), 256,
         1, 1, 1, 0, 0, fmap, 256, 0, 0, ACT_NONE, 1.f, Bcol);

  // cnet on image1 (batch 2), BN-eval; split conv2: rows 0..127 -> tanh(net), 128..255 -> relu into hx ctx
  float* curC = run_encoder(C, image1, nullptr, 2, 1);
  conv2d(stream, curC, 128L * HW, 2, 128, hh, wwq, wh[C.conv2w], bia(C.conv2b), 128,
         1, 1, 1, 0, 0, net, 128, 0, 0, ACT_TANH, 1.f, Bcol);
  conv2d(stream, curC, 128L * HW, 2, 128, hh, wwq, wh[C.conv2w] + 128L * 128,
         bia(C.conv2b) + 128, 128, 1, 1, 1, 0, 0, hx, 384, 128, 0, ACT_RELU, 1.f, Bcol);

  // correlation pyramid: corr = f1^T f2 / 16 via WMMA GEMM, then 2x2 avg pools
  k_fmap2ab<<<eb(2L * 256 * HW), 256, 0, stream>>>(fmap, At, Bc, 256, HW);
  launch_gemm(stream, At, Bc, nullptr, corr0, HW, HW, 256, (long)HW * 256,
              256L * HW, HW, 0, 0, 2, ACT_NONE, 0.0625f);
  k_pool2<<<eb(2L * HW * 560), 256, 0, stream>>>(corr0, corr1, 2 * HW, 40, 56);
  k_pool2<<<eb(2L * HW * 140), 256, 0, stream>>>(corr1, corr2, 2 * HW, 20, 28);
  k_pool2<<<eb(2L * HW * 35), 256, 0, stream>>>(corr2, corr3, 2 * HW, 10, 14);

  k_initcoords<<<eb(2L * 2 * HW), 256, 0, stream>>>(coords, Bn, hh, wwq);

  auto gruDir = [&](int wz, int bz2, int wr, int br2, int wq, int bq2,
                    int KH, int KW, int PH, int PW) {
    int K = 384 * KH * KW;
    long tile = (long)K * HW;
    launch_im2col(stream, hx, 384L * HW, 2, 384, hh, wwq, KH, KW, 1, PH, PW, hh, wwq,
                  Bcol, tile);
    launch_gemm(stream, wh[wz], Bcol, bia(bz2), zb, 128, HW, K, 0, tile, 128, 0, 0, 2,
                ACT_SIG, 1.f);
    launch_gemm(stream, wh[wr], Bcol, bia(br2), rb_, 128, HW, K, 0, tile, 128, 0, 0, 2,
                ACT_SIG, 1.f);
    k_rmul<<<eb(2L * 128 * HW), 256, 0, stream>>>(rb_, net, hx, HW, 2);
    launch_im2col(stream, hx, 384L * HW, 2, 384, hh, wwq, KH, KW, 1, PH, PW, hh, wwq,
                  Bcol, tile);
    launch_gemm(stream, wh[wq], Bcol, bia(bq2), qb, 128, HW, K, 0, tile, 128, 0, 0, 2,
                ACT_TANH, 1.f);
    k_gruupd<<<eb(2L * 128 * HW), 256, 0, stream>>>(zb, qb, net, 2L * 128 * HW);
    k_copych<<<eb(2L * 128 * HW), 256, 0, stream>>>(hx, 384, 0, net, 128, 0, 128, HW, 2);
  };

  for (int it = 0; it < ITERS; ++it) {
    // correlation lookup -> cfeat (f16, direct GEMM-B)
    k_sampler<<<eb(2L * HW * 4 * 81), 256, 0, stream>>>(coords, corr0, corr1, corr2,
                                                        corr3, cfeat, Bn, hh, wwq);
    // motion encoder
    launch_gemm(stream, wh[Ec1w], cfeat, bia(Ec1b), cor1, 256, HW, 324, 0, 324L * HW,
                256, 0, 0, 2, ACT_RELU, 1.f);
    conv2d(stream, cor1, 256L * HW, 2, 256, hh, wwq, wh[Ec2w], bia(Ec2b), 192, 3, 3, 1,
           1, 1, mencat, 256, 0, 0, ACT_RELU, 1.f, Bcol);
    k_flow<<<eb(2L * 2 * HW), 256, 0, stream>>>(coords, flowb, hx, Bn, hh, wwq);
    conv2d(stream, flowb, 2L * HW, 2, 2, hh, wwq, wh[Ef1w], bia(Ef1b), 128, 7, 7, 1,
           3, 3, flo1, 128, 0, 0, ACT_RELU, 1.f, Bcol);
    conv2d(stream, flo1, 128L * HW, 2, 128, hh, wwq, wh[Ef2w], bia(Ef2b), 64, 3, 3, 1,
           1, 1, mencat, 256, 192, 0, ACT_RELU, 1.f, Bcol);
    conv2d(stream, mencat, 256L * HW, 2, 256, hh, wwq, wh[Ecw], bia(Ecb), 126, 3, 3, 1,
           1, 1, hx, 384, 256, 0, ACT_RELU, 1.f, Bcol);
    k_copych<<<eb(2L * 128 * HW), 256, 0, stream>>>(hx, 384, 0, net, 128, 0, 128, HW, 2);
    // SepConvGRU
    gruDir(Gz1w, Gz1b, Gr1w, Gr1b, Gq1w, Gq1b, 1, 5, 0, 2);
    gruDir(Gz2w, Gz2b, Gr2w, Gr2b, Gq2w, Gq2b, 5, 1, 2, 0);
    // flow head
    conv2d(stream, net, 128L * HW, 2, 128, hh, wwq, wh[Fc1w], bia(Fc1b), 256, 3, 3, 1,
           1, 1, fh1, 256, 0, 0, ACT_RELU, 1.f, Bcol);
    conv2d(stream, fh1, 256L * HW, 2, 256, hh, wwq, wh[Fc2w], bia(Fc2b), 2, 3, 3, 1,
           1, 1, dflow, 2, 0, 0, ACT_NONE, 1.f, Bcol);
    k_addinp<<<eb(2L * 2 * HW), 256, 0, stream>>>(coords, dflow, 2L * 2 * HW);
    if (it == ITERS - 1) {  // mask only needed from the last iteration
      conv2d(stream, net, 128L * HW, 2, 128, hh, wwq, wh[Mm1w], bia(Mm1b), 256, 3, 3, 1,
             1, 1, fh1, 256, 0, 0, ACT_RELU, 1.f, Bcol);
      conv2d(stream, fh1, 256L * HW, 2, 256, hh, wwq, wh[Mm2w], bia(Mm2b), 576, 1, 1, 1,
             0, 0, maskb, 576, 0, 0, ACT_NONE, 1.f, Bcol);
    }
  }

  // final flow + convex upsample -> d_out [2,2,320,448]
  k_flow<<<eb(2L * 2 * HW), 256, 0, stream>>>(coords, flowb, nullptr, Bn, hh, wwq);
  k_upsample<<<eb(2L * HW * 64), 256, 0, stream>>>(flowb, maskb, (float*)d_out, Bn, hh, wwq);
  (void)in_sizes; (void)out_size; (void)ws_size;
}
